// GAT_29798483100073
// MI455X (gfx1250) — matrix operand order repaired and checked by reading, not compiled, not run on hardware
//
#include <hip/hip_runtime.h>
#include <hip/hip_bf16.h>

typedef __attribute__((ext_vector_type(16))) _Float16 v16h;
typedef __attribute__((ext_vector_type(8)))  float    v8f;

#define K_DIM  256              // input dim of every layer's GEMM (compile-time -> full unroll)
#define KSTEPS (K_DIM / 32)     // 8 WMMA k-steps
#define HIDC   64
#define NHEADS 4
#define OUTC   64
#define NEG_SLOPE 0.2f

// ---------------------------------------------------------------- utilities

__global__ void k_fill(float* __restrict__ p, float v, int n) {
  int i = blockIdx.x * blockDim.x + threadIdx.x;
  if (i < n) p[i] = v;
}

__global__ void k_deg_accum(const int* __restrict__ dst, float* __restrict__ deg, int E) {
  int i = blockIdx.x * blockDim.x + threadIdx.x;
  if (i < E) atomicAdd(&deg[dst[i]], 1.0f);
}

__device__ __forceinline__ void atomicMaxF(float* a, float v) {
  // sign-aware int-punned float max (lowers to global_atomic_max/min)
  if (v >= 0.0f) atomicMax((int*)a, __float_as_int(v));
  else           atomicMin((unsigned int*)a, __float_as_uint(v));
}

__device__ __forceinline__ void edge_sd(const int* __restrict__ src,
                                        const int* __restrict__ dst,
                                        int e, int E, int& s, int& d) {
  if (e < E) { s = src[e]; d = dst[e]; }
  else       { s = d = e - E; }          // appended self-loops
}

__device__ __forceinline__ float leaky(float v) {
  return v > 0.0f ? v : NEG_SLOPE * v;
}

// ------------------------------------------------- fragment packing kernels
// A fragment layout (16-bit A, 16x32, ISA 7.12.2): element j of lane ->
//   K = j + (j>=8 ? 8 : 0) + (lane>=16 ? 8 : 0),  M = lane & 15
// Packed as [rowTile][kStep][lane][j] so GEMM reads one contiguous 32B per lane.
__global__ void k_pack_a(const float* __restrict__ X, _Float16* __restrict__ Apack, int N) {
  int i = blockIdx.x * blockDim.x + threadIdx.x;
  if (i >= N * K_DIM) return;
  int j    = i & 15;
  int lane = (i >> 4) & 31;
  int t    = i >> 9;              // tm*KSTEPS + ks
  int ks   = t & (KSTEPS - 1);
  int tm   = t >> 3;              // KSTEPS == 8
  int hi = lane >> 4, m = lane & 15;
  int kk = j + (j >= 8 ? 8 : 0) + hi * 8;
  Apack[i] = (_Float16)X[(size_t)(tm * 16 + m) * K_DIM + ks * 32 + kk];
}

// B fragment layout (16-bit B, 32x16): slot j of lane -> K = j + (j>=8 ? 8 : 0) + (lane>=16)*8, N = lane&15.
// Packed as [colTile][kStep][lane][j].
__global__ void k_pack_b(const float* __restrict__ W, _Float16* __restrict__ Bpack, int OUTD) {
  int i = blockIdx.x * blockDim.x + threadIdx.x;
  if (i >= K_DIM * OUTD) return;
  int j    = i & 15;
  int lane = (i >> 4) & 31;
  int t    = i >> 9;              // tn*KSTEPS + ks
  int ks   = t & (KSTEPS - 1);
  int tn   = t >> 3;
  int hi = lane >> 4, nl = lane & 15;
  int k = ks * 32 + j + (j >= 8 ? 8 : 0) + hi * 8;
  Bpack[i] = (_Float16)W[(size_t)k * OUTD + tn * 16 + nl];
}

// ---------------------------------------------------------------- WMMA GEMM
// C[N, OUTD] = A * B from packed f16 fragments, f32 accumulate.
// One wave32 computes a 16x64 strip: 4 col tiles sharing one A fragment.
#define WMMA_F16(a, b, c) \
  __builtin_amdgcn_wmma_f32_16x16x32_f16(false, (a), false, (b), (short)0, (c), false, false)

__global__ void k_gemm_wmma(const _Float16* __restrict__ Apack,
                            const _Float16* __restrict__ Bpack,
                            float* __restrict__ C,
                            int nRowTiles, int OUTD) {
  int groupsN = (OUTD >> 4) >> 2;                 // groups of 4 col tiles
  int wave = (blockIdx.x * blockDim.x + threadIdx.x) >> 5;
  int lane = threadIdx.x & 31;
  if (wave >= nRowTiles * groupsN) return;
  int tm = wave / groupsN;
  int tg = wave - tm * groupsN;

  const _Float16* Ab = Apack + (size_t)tm * KSTEPS * 512 + lane * 16;          // 512 = 32 lanes * 16
  const _Float16* Bb = Bpack + (size_t)(tg * 4) * KSTEPS * 512 + lane * 16;

  v8f acc0 = {}, acc1 = {}, acc2 = {}, acc3 = {};
#pragma unroll
  for (int ks = 0; ks < KSTEPS; ++ks) {
    v16h a  = *(const v16h*)(Ab + (size_t)ks * 512);
    v16h b0 = *(const v16h*)(Bb + (size_t)(0 * KSTEPS + ks) * 512);
    v16h b1 = *(const v16h*)(Bb + (size_t)(1 * KSTEPS + ks) * 512);
    v16h b2 = *(const v16h*)(Bb + (size_t)(2 * KSTEPS + ks) * 512);
    v16h b3 = *(const v16h*)(Bb + (size_t)(3 * KSTEPS + ks) * 512);
    acc0 = WMMA_F16(a, b0, acc0);
    acc1 = WMMA_F16(a, b1, acc1);
    acc2 = WMMA_F16(a, b2, acc2);
    acc3 = WMMA_F16(a, b3, acc3);
  }

  // C/D layout: VGPR r -> M = r + (lane>=16)*8, N = lane & 15
  int hi = lane >> 4, ml = lane & 15;
  float* Cb = C + (size_t)(tm * 16 + hi * 8) * OUTD + tg * 64 + ml;
#pragma unroll
  for (int r = 0; r < 8; ++r) {
    Cb[(size_t)r * OUTD + 0]  = acc0[r];
    Cb[(size_t)r * OUTD + 16] = acc1[r];
    Cb[(size_t)r * OUTD + 32] = acc2[r];
    Cb[(size_t)r * OUTD + 48] = acc3[r];
  }
}

// ---------------------------------------------------------------- attention
__global__ void k_alpha(const float* __restrict__ hlin,
                        const float* __restrict__ avs,
                        const float* __restrict__ avd,
                        float* __restrict__ als, float* __restrict__ ald,
                        int N, int H, int C) {
  int i = blockIdx.x * blockDim.x + threadIdx.x;
  if (i >= N * H) return;
  int n = i / H, h = i - n * H;
  const float* row = hlin + (size_t)n * H * C + (size_t)h * C;
  float ss = 0.0f, sd = 0.0f;
  for (int c = 0; c < C; ++c) {
    float v = row[c];
    ss += v * avs[h * C + c];
    sd += v * avd[h * C + c];
  }
  als[i] = ss;
  ald[i] = sd;
}

// pass 1: per-destination running max of leaky(alpha_s[src]+alpha_d[dst])
__global__ void k_edge_max(const int* __restrict__ src, const int* __restrict__ dst,
                           int E, int N, int H,
                           const float* __restrict__ als, const float* __restrict__ ald,
                           float* __restrict__ mmax) {
  int i = blockIdx.x * blockDim.x + threadIdx.x;
  if (i >= (E + N) * H) return;
  int e = i / H, h = i - e * H;
  int s, d; edge_sd(src, dst, e, E, s, d);
  atomicMaxF(&mmax[d * H + h], leaky(als[s * H + h] + ald[d * H + h]));
}

// pass 2: softmax denominator
__global__ void k_edge_sum(const int* __restrict__ src, const int* __restrict__ dst,
                           int E, int N, int H,
                           const float* __restrict__ als, const float* __restrict__ ald,
                           const float* __restrict__ mmax, float* __restrict__ denom) {
  int i = blockIdx.x * blockDim.x + threadIdx.x;
  if (i >= (E + N) * H) return;
  int e = i / H, h = i - e * H;
  int s, d; edge_sd(src, dst, e, E, s, d);
  float v = leaky(als[s * H + h] + ald[d * H + h]);
  atomicAdd(&denom[d * H + h], __expf(v - mmax[d * H + h]));
}

// pass 3: weighted message scatter-add (one wave per edge, L2-resident atomics)
__global__ void k_edge_scatter(const int* __restrict__ src, const int* __restrict__ dst,
                               int E, int N, int H, int C,
                               const float* __restrict__ als, const float* __restrict__ ald,
                               const float* __restrict__ mmax, const float* __restrict__ denom,
                               const float* __restrict__ hlin, float* __restrict__ accum) {
  int wave = (blockIdx.x * blockDim.x + threadIdx.x) >> 5;
  int lane = threadIdx.x & 31;
  if (wave >= E + N) return;
  int s, d; edge_sd(src, dst, wave, E, s, d);
  int HC = H * C;
  for (int c = lane; c < HC; c += 32) {
    int h = c / C;
    float v = leaky(als[s * H + h] + ald[d * H + h]);
    float w = __expf(v - mmax[d * H + h]) / denom[d * H + h];
    atomicAdd(&accum[(size_t)d * HC + c], hlin[(size_t)s * HC + c] * w);
  }
}

// mean-aggregate + bias + relu -> f32 activations (consumed by k_pack_a)
__global__ void k_finalize_relu(const float* __restrict__ accum, const float* __restrict__ deg,
                                const float* __restrict__ bias, float* __restrict__ actf,
                                int N, int HC) {
  int i = blockIdx.x * blockDim.x + threadIdx.x;
  if (i >= N * HC) return;
  int n = i / HC, c = i - n * HC;
  float v = accum[i] / deg[n] + bias[c];
  actf[i] = v > 0.0f ? v : 0.0f;
}

// last layer: mean-aggregate + bias -> f32 output
__global__ void k_finalize_out(const float* __restrict__ accum, const float* __restrict__ deg,
                               const float* __restrict__ bias, float* __restrict__ out,
                               int N, int HC) {
  int i = blockIdx.x * blockDim.x + threadIdx.x;
  if (i >= N * HC) return;
  int n = i / HC, c = i - n * HC;
  out[i] = accum[i] / deg[n] + bias[c];
}

// ---------------------------------------------------------------- driver

static inline size_t alignUp(size_t x) { return (x + 255) & ~(size_t)255; }

extern "C" void kernel_launch(void* const* d_in, const int* in_sizes, int n_in,
                              void* d_out, int out_size, void* d_ws, size_t ws_size,
                              hipStream_t stream) {
  const int N = in_sizes[0] / K_DIM;       // 50000
  const int E = in_sizes[1] / 2;           // 600000
  const int Etot = E + N;
  const int HC = NHEADS * HIDC;            // 256

  const float* x   = (const float*)d_in[0];
  const int*   ei  = (const int*)d_in[1];
  const int*   src = ei;
  const int*   dst = ei + E;

  const float* W[3]  = { (const float*)d_in[2], (const float*)d_in[6],  (const float*)d_in[10] };
  const float* As[3] = { (const float*)d_in[3], (const float*)d_in[7],  (const float*)d_in[11] };
  const float* Ad[3] = { (const float*)d_in[4], (const float*)d_in[8],  (const float*)d_in[12] };
  const float* Bb[3] = { (const float*)d_in[5], (const float*)d_in[9],  (const float*)d_in[13] };
  const int    Hh[3] = { NHEADS, NHEADS, 1 };
  const int    Cc[3] = { HIDC,   HIDC,   OUTC };

  // ---- workspace carve (feature matrices are L2-resident: 51MB < 192MB L2) ----
  char* p = (char*)d_ws;
  size_t off = 0;
  auto carve = [&](size_t bytes) { void* q = p + off; off = alignUp(off + bytes); return q; };
  _Float16* Apack = (_Float16*)carve((size_t)N * K_DIM * sizeof(_Float16));     // packed activations
  _Float16* Bpack = (_Float16*)carve((size_t)K_DIM * HC * sizeof(_Float16));    // packed weights
  float*    hlin  = (float*)carve((size_t)N * HC * sizeof(float));  // GEMM out, then next-layer f32 act
  float*    accum = (float*)carve((size_t)N * HC * sizeof(float));
  float*    als   = (float*)carve((size_t)N * NHEADS * sizeof(float));
  float*    ald   = (float*)carve((size_t)N * NHEADS * sizeof(float));
  float*    mmax  = (float*)carve((size_t)N * NHEADS * sizeof(float));
  float*    denom = (float*)carve((size_t)N * NHEADS * sizeof(float));
  float*    deg   = (float*)carve((size_t)N * sizeof(float));
  (void)ws_size;

  const int T = 256;
  auto nb = [&](long long n) { return (int)((n + T - 1) / T); };

  // degrees: self-loop contributes 1, then count real incoming edges
  k_fill<<<nb(N), T, 0, stream>>>(deg, 1.0f, N);
  k_deg_accum<<<nb(E), T, 0, stream>>>(dst, deg, E);

  for (int l = 0; l < 3; ++l) {
    const int H = Hh[l], C = Cc[l];
    const int OUTD = H * C;                       // 256, 256, 64
    const float* actIn = (l == 0) ? x : hlin;     // f32 activations from previous finalize

    // pack weights + activations into WMMA fragment order (b128-friendly)
    k_pack_b<<<nb((long long)K_DIM * OUTD), T, 0, stream>>>(W[l], Bpack, OUTD);
    k_pack_a<<<nb((long long)N * K_DIM), T, 0, stream>>>(actIn, Apack, N);

    // dense GEMM on the matrix pipe: 32x v_wmma_f32_16x16x32_f16 per wave
    {
      long long waves = (long long)(N / 16) * ((OUTD / 16) / 4);
      k_gemm_wmma<<<(int)((waves + 7) / 8), T, 0, stream>>>(Apack, Bpack, hlin, N / 16, OUTD);
    }

    // attention logits
    k_alpha<<<nb((long long)N * H), T, 0, stream>>>(hlin, As[l], Ad[l], als, ald, N, H, C);

    // per-layer reductions re-initialized every call (graph-replay safe)
    k_fill<<<nb((long long)N * H), T, 0, stream>>>(mmax, -1.0e30f, N * H);
    k_fill<<<nb((long long)N * H), T, 0, stream>>>(denom, 0.0f, N * H);
    k_fill<<<nb((long long)N * OUTD), T, 0, stream>>>(accum, 0.0f, N * OUTD);

    // segment softmax + aggregation over edges (incl. self-loops)
    k_edge_max<<<nb((long long)Etot * H), T, 0, stream>>>(src, dst, E, N, H, als, ald, mmax);
    k_edge_sum<<<nb((long long)Etot * H), T, 0, stream>>>(src, dst, E, N, H, als, ald, mmax, denom);
    k_edge_scatter<<<(Etot + 7) / 8, T, 0, stream>>>(src, dst, E, N, H, C,
                                                     als, ald, mmax, denom, hlin, accum);

    if (l < 2) {  // mean + bias + relu -> f32 act for next layer (hlin reused)
      k_finalize_relu<<<nb((long long)N * OUTD), T, 0, stream>>>(accum, deg, Bb[l], hlin, N, OUTD);
    } else {
      k_finalize_out<<<nb((long long)N * OUTD), T, 0, stream>>>(accum, deg, Bb[l], (float*)d_out, N, OUTD);
    }
  }
}
